// NonLocal2d_89154931131022
// MI455X (gfx1250) — matrix operand order repaired and checked by reading, compiled, not run on hardware
//
#include <hip/hip_runtime.h>

typedef __attribute__((ext_vector_type(2)))  _Float16 v2h;
typedef __attribute__((ext_vector_type(2)))  __fp16   v2hp;   // cvt_pkrtz result type
typedef __attribute__((ext_vector_type(4)))  _Float16 v4h;
typedef __attribute__((ext_vector_type(8)))  _Float16 v8h;
typedef __attribute__((ext_vector_type(16))) _Float16 v16h;
typedef __attribute__((ext_vector_type(8)))  float    v8f;

#define B_   8
#define C_   64
#define N_   6400
#define BR_  128            // query rows per block (8 waves x 16)
#define BC_  64             // key columns per iteration
#define NJ_  (N_ / BC_)     // 100
#define NRB_ (N_ / BR_)     // 50
#define NT_  (N_ / 64)      // 100 n-tiles for the convert pass
#define WAVES_ 8

#define WMMA_F16(A, B, Cacc) \
  __builtin_amdgcn_wmma_f32_16x16x32_f16(false, (A), false, (B), (short)0, (Cacc), false, false)

__device__ __forceinline__ unsigned pk_max_f16x2(unsigned a, unsigned b) {
  unsigned d;
  asm("v_pk_max_num_f16 %0, %1, %2" : "=v"(d) : "v"(a), "v"(b));
  return d;
}

// ---------------------------------------------------------------------------
// Pre-pass: x (f32, C x N per batch) -> x16 (f16, same layout) and
//           xT16 (f16, N x C transposed layout), via an LDS tile transpose.
// ---------------------------------------------------------------------------
__global__ __launch_bounds__(256) void convert_kernel(
    const float* __restrict__ x, _Float16* __restrict__ x16,
    _Float16* __restrict__ xT16) {
  __shared__ _Float16 tile[64][64];
  const int tid = threadIdx.x;
  const int blk = blockIdx.x;
  const int b   = blk / NT_;
  const int n0  = (blk % NT_) * 64;

  const float* xb  = x    + (size_t)b * C_ * N_;
  _Float16* x16b   = x16  + (size_t)b * C_ * N_;
  _Float16* xT16b  = xT16 + (size_t)b * N_ * C_;

#pragma unroll
  for (int pass = 0; pass < 4; ++pass) {
    const int c  = pass * 16 + (tid >> 4);
    const int n4 = (tid & 15) * 4;
    const float4 v = *(const float4*)(xb + (size_t)c * N_ + n0 + n4);
    v4h h4;
    h4[0] = (_Float16)v.x; h4[1] = (_Float16)v.y;
    h4[2] = (_Float16)v.z; h4[3] = (_Float16)v.w;
    *(v4h*)(x16b + (size_t)c * N_ + n0 + n4) = h4;   // coalesced f16 copy
    tile[c][n4 + 0] = h4[0]; tile[c][n4 + 1] = h4[1];
    tile[c][n4 + 2] = h4[2]; tile[c][n4 + 3] = h4[3];
  }
  __syncthreads();

  // each thread emits 32 contiguous bytes of one transposed row
  const int n  = tid >> 2;
  const int cq = (tid & 3) * 16;
  v16h r;
#pragma unroll
  for (int i = 0; i < 16; ++i) r[i] = tile[cq + i][n];
  *(v16h*)(xT16b + (size_t)(n0 + n) * C_ + cq) = r;
}

// ---------------------------------------------------------------------------
// Main fused flash-attention kernel (v2): all WMMA fragments loaded straight
// from global f16 copies (32B contiguous per lane); no staging, no workgroup
// barriers in the key loop. Fragments held in explicit register arrays so
// loads issue as clauses and overlap with WMMA / softmax VALU.
// ---------------------------------------------------------------------------
__global__ __launch_bounds__(256) void nonlocal_attn_v2(
    const float* __restrict__ x, const _Float16* __restrict__ x16,
    const _Float16* __restrict__ xT16, float* __restrict__ out) {

  __shared__ alignas(32) _Float16 pfrag[WAVES_][2][32][16];  // 16 KB, per-wave

  const int tid  = threadIdx.x;
  const int wave = tid >> 5;
  const int lane = tid & 31;
  const int half = lane >> 4;
  const int colL = lane & 15;

  // scalar (SGPR) copy of the wave id -> uniform branch, no exec masking
  const bool wave0 = (__builtin_amdgcn_readfirstlane(wave) == 0);

  const int blk  = blockIdx.x;
  const int b    = blk / NRB_;
  const int rb   = blk % NRB_;
  const int row0 = rb * BR_;

  const float*    xb    = x    + (size_t)b * C_ * N_;
  const _Float16* x16b  = x16  + (size_t)b * C_ * N_;
  const _Float16* xT16b = xT16 + (size_t)b * N_ * C_;

  // ---- Q A-fragments straight from xT16 (two 16B chunks per fragment) ----
  // A k order chosen here (not the ISA table order): lanes<16 hold K 0..15; lanes>=16 hold K 16..31 (contiguous).
  const int mrow = row0 + wave * 16 + colL;
  const _Float16* qrow = xT16b + (size_t)mrow * C_;
  v16h qa[2];
#pragma unroll
  for (int h = 0; h < 2; ++h) {
    const int cb = h * 32 + (half ? 16 : 0);
    const v8h lo = *(const v8h*)(qrow + cb);
    const v8h hi = *(const v8h*)(qrow + cb + 8);
    qa[h] = __builtin_shufflevector(lo, hi, 0, 1, 2, 3, 4, 5, 6, 7,
                                    8, 9, 10, 11, 12, 13, 14, 15);
  }

  v16h ones;
#pragma unroll
  for (int i = 0; i < 16; ++i) ones[i] = (_Float16)1.0f;

  float rmax[8], rsum[8];
  v8f   oacc[4] = {};
#pragma unroll
  for (int r = 0; r < 8; ++r) { rmax[r] = -__builtin_inff(); rsum[r] = 0.0f; }

  // B k order chosen here to match A: lanes<16 hold K 0..15, lanes>=16 hold K 16..31 (contiguous)
  const int koff = half ? 16 : 0;

  for (int jb = 0; jb < NJ_; ++jb) {
    const int j0 = jb * BC_;

    // ---- next-block prefetch: one uniform branch, all 8 lines at once ----
    if (wave0 && jb + 1 < NJ_) {
      const _Float16* nk = xT16b + (size_t)(j0 + BC_ + colL) * C_;
      const _Float16* nv = x16b + (size_t)colL * N_ + j0 + BC_;
#pragma unroll
      for (int g = 0; g < 4; ++g) {
        __builtin_prefetch(nk + (size_t)g * 16 * C_, 0, 3);
        __builtin_prefetch(nv + (size_t)g * 16 * N_, 0, 3);
      }
    }

    // ---- issue ALL K-fragment loads up front (distinct registers) ----
    v16h kb[4][2];
#pragma unroll
    for (int g = 0; g < 4; ++g) {
      const _Float16* kc = xT16b + (size_t)(j0 + g * 16 + colL) * C_;
      kb[g][0] = *(const v16h*)(kc + koff);
      kb[g][1] = *(const v16h*)(kc + 32 + koff);
    }

    // ---- S = Q K^T : 16x64 per wave, f32 accum ----
    v8f s[4];
#pragma unroll
    for (int g = 0; g < 4; ++g) {
      v8f acc = {};
      acc  = WMMA_F16(qa[0], kb[g][0], acc);
      acc  = WMMA_F16(qa[1], kb[g][1], acc);
      s[g] = acc;
    }

    // ---- V-fragment loads issued now; softmax VALU below hides latency ----
    v16h vb[4][2];
#pragma unroll
    for (int g = 0; g < 4; ++g) {
      const _Float16* vc = x16b + (size_t)(g * 16 + colL) * N_ + j0;
      vb[g][0] = *(const v16h*)(vc + koff);
      vb[g][1] = *(const v16h*)(vc + 32 + koff);
    }

    // ---- online softmax: row max over 64 columns ----
    // f32 max across the 4 column groups, then pack row pairs to f16x2 and
    // reduce across the 16 column lanes with VOP3P packed max (RTZ pack under-
    // estimates the max by <1 f16 ulp -> exp(s-m) <= e^0.03, still f16-safe).
    float mx[8];
#pragma unroll
    for (int r = 0; r < 8; ++r)
      mx[r] = fmaxf(fmaxf(s[0][r], s[1][r]), fmaxf(s[2][r], s[3][r]));
    unsigned pk[4];
#pragma unroll
    for (int i = 0; i < 4; ++i) {
      const v2hp p2 = __builtin_amdgcn_cvt_pkrtz(mx[2 * i], mx[2 * i + 1]);
      pk[i] = __builtin_bit_cast(unsigned, p2);
    }
#pragma unroll
    for (int msk = 1; msk <= 8; msk <<= 1) {
#pragma unroll
      for (int i = 0; i < 4; ++i)
        pk[i] = pk_max_f16x2(pk[i], (unsigned)__shfl_xor(pk[i], msk, 32));
    }
    float al[8];
#pragma unroll
    for (int i = 0; i < 4; ++i) {
      const v2h p2 = __builtin_bit_cast(v2h, pk[i]);
      const float m0 = fmaxf(rmax[2 * i],     (float)p2[0]);
      const float m1 = fmaxf(rmax[2 * i + 1], (float)p2[1]);
      al[2 * i]     = __expf(rmax[2 * i] - m0);          // first iter: 0
      al[2 * i + 1] = __expf(rmax[2 * i + 1] - m1);
      rmax[2 * i] = m0; rmax[2 * i + 1] = m1;
    }
#pragma unroll
    for (int g = 0; g < 4; ++g)
#pragma unroll
      for (int r = 0; r < 8; ++r) oacc[g][r] *= al[r];

    // ---- P = exp(S - max) -> per-wave A-fragment transpose through LDS ----
#pragma unroll
    for (int g = 0; g < 4; ++g) {
#pragma unroll
      for (int r = 0; r < 8; ++r) {
        const float pv = __expf(s[g][r] - rmax[r]);
        const int n  = g * 16 + colL;
        const int h  = n >> 5, nn = n & 31;
        const int p  = nn & 15;
        const int lh = nn >> 4;
        pfrag[wave][h][(r + 8 * half) + 16 * lh][p] = (_Float16)pv;
      }
    }
    __builtin_amdgcn_wave_barrier();
    asm volatile("s_wait_dscnt 0" ::: "memory");

    const v16h pa0 = *(const v16h*)&pfrag[wave][0][lane][0];
    const v16h pa1 = *(const v16h*)&pfrag[wave][1][lane][0];

    // ---- row sums via WMMA: rs = P * ones (replicated across columns) ----
    v8f rs = {};
    rs = WMMA_F16(pa0, ones, rs);
    rs = WMMA_F16(pa1, ones, rs);
#pragma unroll
    for (int r = 0; r < 8; ++r) rsum[r] = fmaf(rsum[r], al[r], rs[r]);

    // ---- O += P V : V-fragments already resident ----
#pragma unroll
    for (int g = 0; g < 4; ++g) {
      oacc[g] = WMMA_F16(pa0, vb[g][0], oacc[g]);
      oacc[g] = WMMA_F16(pa1, vb[g][1], oacc[g]);
    }
  }

  // ---- epilogue: out[b,c,n] = O[m,c] / l[m] + x[b,c,n] ----
  float inv[8];
#pragma unroll
  for (int r = 0; r < 8; ++r) inv[r] = 1.0f / rsum[r];

  float* ob = out + (size_t)b * C_ * N_;
#pragma unroll
  for (int g = 0; g < 4; ++g) {
#pragma unroll
    for (int r = 0; r < 8; ++r) {
      const int c = g * 16 + colL;
      const int m = row0 + wave * 16 + r + 8 * half;
      const size_t idx = (size_t)c * N_ + m;
      ob[idx] = oacc[g][r] * inv[r] + xb[idx];
    }
  }
}

// ---------------------------------------------------------------------------
// Fallback (round-1 kernel): fully fused, LDS staging, no workspace needed.
// ---------------------------------------------------------------------------
__global__ __launch_bounds__(256) void nonlocal_attn_v1(
    const float* __restrict__ x, float* __restrict__ out) {

  __shared__ alignas(32) _Float16 qfrag[WAVES_][2][32][16];
  __shared__ alignas(32) _Float16 kfrag[4][2][32][16];
  __shared__ alignas(32) _Float16 vfrag[4][2][32][16];
  __shared__ alignas(32) _Float16 pfrag[WAVES_][2][32][16];

  const int tid  = threadIdx.x;
  const int wave = tid >> 5;
  const int lane = tid & 31;
  const int half = lane >> 4;
  const int colL = lane & 15;

  const int blk  = blockIdx.x;
  const int b    = blk / NRB_;
  const int rb   = blk % NRB_;
  const int row0 = rb * BR_;

  const float* xb = x + (size_t)b * C_ * N_;

#pragma unroll
  for (int pass = 0; pass < 8; ++pass) {
    const int c  = pass * 8 + (tid >> 5);
    const int m0 = (tid & 31) * 4;
    const float4 v = *(const float4*)(xb + (size_t)c * N_ + row0 + m0);
    const float vv[4] = {v.x, v.y, v.z, v.w};
#pragma unroll
    for (int e = 0; e < 4; ++e) {
      const int m  = m0 + e;
      const int w  = m >> 4, mm = m & 15;
      const int h  = c >> 5, cc = c & 31;
      const int p  = cc & 15;
      const int lh = cc >> 4;
      qfrag[w][h][mm + 16 * lh][p] = (_Float16)vv[e];
    }
  }
  __syncthreads();

  const v16h qa0 = *(const v16h*)&qfrag[wave][0][lane][0];
  const v16h qa1 = *(const v16h*)&qfrag[wave][1][lane][0];

  float rmax[8], rsum[8];
  v8f   oacc[4] = {};
#pragma unroll
  for (int r = 0; r < 8; ++r) { rmax[r] = -__builtin_inff(); rsum[r] = 0.0f; }

  for (int jb = 0; jb < NJ_; ++jb) {
    const int j0 = jb * BC_;
    __syncthreads();
#pragma unroll
    for (int pass = 0; pass < 4; ++pass) {
      const int c  = pass * 16 + (tid >> 4);
      const int n4 = (tid & 15) * 4;
      const float4 v = *(const float4*)(xb + (size_t)c * N_ + j0 + n4);
      if (jb + 1 < NJ_)
        __builtin_prefetch(xb + (size_t)c * N_ + j0 + BC_ + n4, 0, 3);
      const float vv[4] = {v.x, v.y, v.z, v.w};
#pragma unroll
      for (int e = 0; e < 4; ++e) {
        const int n = n4 + e;
        const _Float16 f = (_Float16)vv[e];
        {
          const int g = n >> 4, nn = n & 15;
          const int h = c >> 5, cc = c & 31;
          kfrag[g][h][nn + 16 * (cc >> 4)][cc & 15] = f;
        }
        {
          const int gc = c >> 4, cl = c & 15;
          const int hm = n >> 5, mm = n & 31;
          vfrag[gc][hm][cl + 16 * (mm >> 4)][mm & 15] = f;
        }
      }
    }
    __syncthreads();

    v8f s[4];
#pragma unroll
    for (int g = 0; g < 4; ++g) {
      const v16h kb0 = *(const v16h*)&kfrag[g][0][lane][0];
      const v16h kb1 = *(const v16h*)&kfrag[g][1][lane][0];
      v8f acc = {};
      acc  = WMMA_F16(qa0, kb0, acc);
      acc  = WMMA_F16(qa1, kb1, acc);
      s[g] = acc;
    }

    float mx[8];
#pragma unroll
    for (int r = 0; r < 8; ++r) {
      float m = fmaxf(fmaxf(s[0][r], s[1][r]), fmaxf(s[2][r], s[3][r]));
      m = fmaxf(m, __shfl_xor(m, 1, 32));
      m = fmaxf(m, __shfl_xor(m, 2, 32));
      m = fmaxf(m, __shfl_xor(m, 4, 32));
      m = fmaxf(m, __shfl_xor(m, 8, 32));
      mx[r] = fmaxf(rmax[r], m);
    }
    float al[8];
#pragma unroll
    for (int r = 0; r < 8; ++r) {
      al[r]   = __expf(rmax[r] - mx[r]);
      rmax[r] = mx[r];
      rsum[r] *= al[r];
    }
#pragma unroll
    for (int g = 0; g < 4; ++g)
#pragma unroll
      for (int r = 0; r < 8; ++r) oacc[g][r] *= al[r];

    float ps[8] = {};
#pragma unroll
    for (int g = 0; g < 4; ++g) {
#pragma unroll
      for (int r = 0; r < 8; ++r) {
        const float pv = __expf(s[g][r] - rmax[r]);
        ps[r] += pv;
        const int n  = g * 16 + colL;
        const int h  = n >> 5, nn = n & 31;
        const int p  = nn & 15;
        const int lh = nn >> 4;
        pfrag[wave][h][(r + 8 * half) + 16 * lh][p] = (_Float16)pv;
      }
    }
#pragma unroll
    for (int r = 0; r < 8; ++r) {
      float t = ps[r];
      t += __shfl_xor(t, 1, 32);
      t += __shfl_xor(t, 2, 32);
      t += __shfl_xor(t, 4, 32);
      t += __shfl_xor(t, 8, 32);
      rsum[r] += t;
    }

    __builtin_amdgcn_wave_barrier();
    asm volatile("s_wait_dscnt 0" ::: "memory");

    const v16h pa0 = *(const v16h*)&pfrag[wave][0][lane][0];
    const v16h pa1 = *(const v16h*)&pfrag[wave][1][lane][0];

#pragma unroll
    for (int g = 0; g < 4; ++g) {
      const v16h vb0 = *(const v16h*)&vfrag[g][0][lane][0];
      const v16h vb1 = *(const v16h*)&vfrag[g][1][lane][0];
      oacc[g] = WMMA_F16(pa0, vb0, oacc[g]);
      oacc[g] = WMMA_F16(pa1, vb1, oacc[g]);
    }
  }

  float inv[8];
#pragma unroll
  for (int r = 0; r < 8; ++r) inv[r] = 1.0f / rsum[r];

  float* ob = out + (size_t)b * C_ * N_;
#pragma unroll
  for (int g = 0; g < 4; ++g) {
#pragma unroll
    for (int r = 0; r < 8; ++r) {
      const int c = g * 16 + colL;
      const int m = row0 + wave * 16 + r + 8 * half;
      const size_t idx = (size_t)c * N_ + m;
      ob[idx] = oacc[g][r] * inv[r] + xb[idx];
    }
  }
}

extern "C" void kernel_launch(void* const* d_in, const int* in_sizes, int n_in,
                              void* d_out, int out_size, void* d_ws, size_t ws_size,
                              hipStream_t stream) {
  (void)in_sizes; (void)n_in; (void)out_size;
  const float* x = (const float*)d_in[0];
  float* out = (float*)d_out;

  const size_t elems = (size_t)B_ * C_ * N_;
  const size_t need  = 2 * elems * sizeof(_Float16);   // x16 + xT16

  if (ws_size >= need) {
    _Float16* x16  = (_Float16*)d_ws;
    _Float16* xT16 = x16 + elems;
    convert_kernel<<<dim3(B_ * NT_), 256, 0, stream>>>(x, x16, xT16);
    nonlocal_attn_v2<<<dim3(B_ * NRB_), 256, 0, stream>>>(x, x16, xT16, out);
  } else {
    nonlocal_attn_v1<<<dim3(B_ * NRB_), 256, 0, stream>>>(x, out);
  }
}
